// Renderer_8272107012721
// MI455X (gfx1250) — compile-verified, matrix operand order checked by reading, not run on hardware
//
#include <hip/hip_runtime.h>

// ============================================================================
// MI455X / gfx1250 implementation of the IBRNet-style renderer.
//
// Assumptions (documented, compile-only environment):
//  * d_in order = setup_inputs() insertion order, params dict flattened in
//    insertion order: [viewdirs, feat, occ_masks, tok_w, tok_b, vm1_w, vm1_b,
//    vm2_w, vm2_b, 4x{wq,wk,wv,fc,ln1_g,ln1_b,w1,b1,w2,b2,ln2_g,ln2_b},
//    c1_w,c1_b,c2_w,c2_b,c3_w,c3_b,t1_w,t1_b,t2_w,t2_b,t3_w,t3_b,co_w,co_b,
//    sig1_w..sig3_b, rgb1_w..rgb3_b]  (83 leaves).
//  * ws_size >= ~282 MB (token state 151MB + masks + autoencoder buffers).
//  * WMMA B-fragment layout mirrors the A layout with N-per-lane (column-major
//    K x N == row-major W[n][k]), per CDNA5 ISA 7.12.2 conventions.
// ============================================================================

#define N_RAYSC 1024
#define S_SAMP  128
#define BTOT    (N_RAYSC * S_SAMP)   // 131072 samples
#define NV      8                    // views
#define DDIM    32

typedef _Float16 half_t;
typedef __attribute__((ext_vector_type(16))) _Float16 v16h;
typedef __attribute__((ext_vector_type(8)))  _Float16 v8h;
typedef __attribute__((ext_vector_type(8)))  float    v8f;

__device__ __forceinline__ float elu_f(float x) { return x > 0.f ? x : (__expf(x) - 1.f); }

// ---------------------------------------------------------------------------
// WMMA fragment helpers (V_WMMA_F32_16X16X32_F16, wave32).
// A (16x32 f16): lane<16 -> M=lane, halves [0..7]=K0..7, [8..15]=K16..23;
//                lane>=16 -> M=lane-16, halves [0..7]=K8..15, [8..15]=K24..31.
// => k = h + (h>=8 ? 8 : 0) + 8*lane[4].   B uses the same map with N per lane.
// C/D (16x16 f32, 8 VGPRs): n = lane&15, m = reg + 8*lane[4].
// ---------------------------------------------------------------------------
__device__ __forceinline__ v16h frag_f32(const float* __restrict__ base, int lane) {
  const int r = lane & 15, khi = (lane >> 4) & 1;
  const float* row = base + r * 32 + khi * 8;
  v16h f;
#pragma unroll
  for (int h = 0; h < 8; ++h)  f[h] = (_Float16)row[h];
#pragma unroll
  for (int h = 8; h < 16; ++h) f[h] = (_Float16)row[h + 8];
  return f;
}
__device__ __forceinline__ v16h frag_f16(const half_t* __restrict__ base, int lane) {
  const int r = lane & 15, khi = (lane >> 4) & 1;
  const half_t* row = base + r * 32 + khi * 8;
  v16h f;
#pragma unroll
  for (int h = 0; h < 8; ++h)  f[h] = row[h];
#pragma unroll
  for (int h = 8; h < 16; ++h) f[h] = row[h + 8];
  return f;
}
// B fragment from a row-major (32 out, 32 in) f32 weight matrix (x @ W^T).
__device__ __forceinline__ v16h fragB(const float* __restrict__ w, int n0, int lane) {
  const int r = lane & 15, khi = (lane >> 4) & 1;
  const float* row = w + (size_t)(n0 + r) * 32 + khi * 8;
  v16h f;
#pragma unroll
  for (int h = 0; h < 8; ++h)  f[h] = (_Float16)row[h];
#pragma unroll
  for (int h = 8; h < 16; ++h) f[h] = (_Float16)row[h + 8];
  return f;
}

// 144x32 (A fragments pre-built in registers) @ W(32x32)^T -> f16 dst
__device__ __forceinline__ void mm_frags_to_half(const v16h* __restrict__ afr,
                                                 const float* __restrict__ W,
                                                 half_t* __restrict__ dst, int lane) {
  const int n = lane & 15, mb = ((lane >> 4) & 1) * 8;
  v8f zero = {0.f, 0.f, 0.f, 0.f, 0.f, 0.f, 0.f, 0.f};
  v16h bn0 = fragB(W, 0, lane);
  v16h bn1 = fragB(W, 16, lane);
#pragma unroll
  for (int t = 0; t < 9; ++t) {
    v8f c0 = __builtin_amdgcn_wmma_f32_16x16x32_f16(false, afr[t], false, bn0, (short)0, zero, false, false);
    v8f c1 = __builtin_amdgcn_wmma_f32_16x16x32_f16(false, afr[t], false, bn1, (short)0, zero, false, false);
#pragma unroll
    for (int r = 0; r < 8; ++r) {
      const int row = t * 16 + mb + r;
      dst[row * 32 + n]      = (half_t)c0[r];
      dst[row * 32 + 16 + n] = (half_t)c1[r];
    }
  }
}

// ---------------------------------------------------------------------------
// Kernel 1: token embedding (8 view tokens + vm token) and visibility masks.
// ---------------------------------------------------------------------------
__global__ __launch_bounds__(256) void embed_kernel(
    const float* __restrict__ feat, const float* __restrict__ occ,
    const float* __restrict__ tokw, const float* __restrict__ tokb,
    const float* __restrict__ vm1w, const float* __restrict__ vm1b,
    const float* __restrict__ vm2w, const float* __restrict__ vm2b,
    float* __restrict__ tok, float* __restrict__ masks)
{
  const int b = blockIdx.x * blockDim.x + threadIdx.x;
  if (b >= BTOT) return;
  const float* f = feat + (size_t)b * NV * 36;
  float* trow = tok + (size_t)b * 9 * 32;

  // variance / mean of s_feat over views
  float mn[8], m2[8];
#pragma unroll
  for (int d = 0; d < 8; ++d) { mn[d] = 0.f; m2[d] = 0.f; }
  for (int v = 0; v < NV; ++v)
#pragma unroll
    for (int d = 0; d < 8; ++d) { float s = f[v * 36 + 24 + d]; mn[d] += s; m2[d] += s * s; }
  float vmin[16];
#pragma unroll
  for (int d = 0; d < 8; ++d) {
    float m = mn[d] * 0.125f;
    vmin[8 + d] = m;
    vmin[d] = m2[d] * 0.125f - m * m;
  }
  float h1[32];
  for (int o = 0; o < 32; ++o) {
    float acc = vm1b[o];
    const float* wr = vm1w + o * 16;
    for (int i = 0; i < 16; ++i) acc = fmaf(wr[i], vmin[i], acc);
    h1[o] = elu_f(acc);
  }
  for (int o = 0; o < 32; ++o) {
    float acc = vm2b[o];
    const float* wr = vm2w + o * 32;
    for (int i = 0; i < 32; ++i) acc = fmaf(wr[i], h1[i], acc);
    trow[8 * 32 + o] = elu_f(acc);   // vm token = row 8
  }
  // per-view tokens: concat(v_feat[24], vis[1], s_feat[8]) -> 33 -> 32, elu
  for (int v = 0; v < NV; ++v) {
    const float* fv = f + v * 36;
    float x[33];
#pragma unroll
    for (int i = 0; i < 24; ++i) x[i] = fv[i];
    x[24] = fv[35];
#pragma unroll
    for (int i = 0; i < 8; ++i) x[25 + i] = fv[24 + i];
    for (int o = 0; o < 32; ++o) {
      float acc = tokb[o];
      const float* wr = tokw + o * 33;
      for (int i = 0; i < 33; ++i) acc = fmaf(wr[i], x[i], acc);
      trow[v * 32 + o] = elu_f(acc);
    }
  }
  // masks
  float v9[9], mc[9];
#pragma unroll
  for (int v = 0; v < 8; ++v) v9[v] = f[v * 36 + 35];
  v9[8] = 1.f;
  float s1 = 0.f;
#pragma unroll
  for (int k = 0; k < 9; ++k) s1 += v9[k];
  if (s1 == 1.f) {
#pragma unroll
    for (int k = 0; k < 9; ++k) v9[k] = 1.f;
  }
#pragma unroll
  for (int k = 0; k < 9; ++k) mc[k] = v9[k];
  const float* oc = occ + (size_t)b * 8;
#pragma unroll
  for (int v = 0; v < 8; ++v) v9[v] *= oc[v];
  float s2 = 0.f;
#pragma unroll
  for (int k = 0; k < 9; ++k) s2 += v9[k];
  if (s2 == 1.f) {
#pragma unroll
    for (int k = 0; k < 9; ++k) v9[k] = 1.f;
  }
  float* mrow = masks + (size_t)b * 9;
#pragma unroll
  for (int k = 0; k < 9; ++k) mrow[k] = v9[k] * mc[k];
}

// ---------------------------------------------------------------------------
// Kernel 2: one transformer encoder layer.  1 wave per block, 16 samples per
// wave (144 token rows = 9 WMMA M-tiles).  Tokens enter/leave LDS via the
// gfx1250 async LDS path; all GEMMs on v_wmma_f32_16x16x32_f16.
// ---------------------------------------------------------------------------
__global__ __launch_bounds__(32) void xf_layer(
    float* __restrict__ tok, const float* __restrict__ masks,
    const float* __restrict__ wq, const float* __restrict__ wk,
    const float* __restrict__ wv, const float* __restrict__ fcw,
    const float* __restrict__ ln1g, const float* __restrict__ ln1b,
    const float* __restrict__ w1, const float* __restrict__ bb1,
    const float* __restrict__ w2, const float* __restrict__ bb2,
    const float* __restrict__ ln2g, const float* __restrict__ ln2b)
{
  __shared__ __attribute__((aligned(16))) float  tokf[144 * 32];  // f32 tokens / x / y
  __shared__ __attribute__((aligned(16))) half_t bq[144 * 32];    // Q -> attn out O
  __shared__ __attribute__((aligned(16))) half_t bk[144 * 32];    // K -> FFN hidden
  __shared__ __attribute__((aligned(16))) half_t bv[144 * 32];    // V

  const int lane = threadIdx.x;
  const int n    = lane & 15;
  const int mb   = ((lane >> 4) & 1) * 8;
  const size_t gb0 = (size_t)blockIdx.x * 16;
  float* gtok = tok + gb0 * 9 * 32;
  v8f zero = {0.f, 0.f, 0.f, 0.f, 0.f, 0.f, 0.f, 0.f};

  // ---- async copy 144x32 f32 tokens global -> LDS (ASYNCcnt tracked) ----
  {
    const unsigned ldsb = (unsigned)(size_t)tokf;  // low 32 bits of generic = LDS addr
    const unsigned long long gbase = (unsigned long long)(size_t)gtok;
    for (int i = lane; i < (144 * 32) / 4; i += 32) {
      unsigned la = ldsb + (unsigned)i * 16u;
      unsigned long long ga = gbase + (unsigned long long)i * 16ull;
      asm volatile("global_load_async_to_lds_b128 %0, %1, off"
                   :: "v"(la), "v"(ga) : "memory");
    }
    asm volatile("s_wait_asynccnt 0" ::: "memory");
  }

  // ---- build the 9 A-fragments once; reuse for Q, K, V projections ----
  v16h afr[9];
#pragma unroll
  for (int t = 0; t < 9; ++t) afr[t] = frag_f32(tokf + t * 512, lane);
  mm_frags_to_half(afr, wq, bq, lane);
  mm_frags_to_half(afr, wk, bk, lane);
  mm_frags_to_half(afr, wv, bv, lane);

  // ---- attention: 64 (sample, head) units over 32 lanes, 2 per lane ----
  for (int j = 0; j < 2; ++j) {
    const int u  = lane + 32 * j;   // 0..63
    const int bb = u >> 2;          // local sample 0..15
    const int h  = u & 3;           // head
    // rows are 8 contiguous 16B-aligned halves -> single ds_load_b128 each
    const v8h* Qv = (const v8h*)(bq + (bb * 9) * 32 + h * 8);  // stride 4 (v8h units)
    const v8h* Kv = (const v8h*)(bk + (bb * 9) * 32 + h * 8);
    const v8h* Vv = (const v8h*)(bv + (bb * 9) * 32 + h * 8);
    float msk[9];
#pragma unroll
    for (int k2 = 0; k2 < 9; ++k2) msk[k2] = masks[(gb0 + bb) * 9 + k2];
    for (int q = 0; q < 9; ++q) {
      const v8h qr = Qv[q * 4];
      float qv[8];
#pragma unroll
      for (int d = 0; d < 8; ++d) qv[d] = (float)qr[d];
      float sr[9], mx = -1e30f;
#pragma unroll
      for (int k2 = 0; k2 < 9; ++k2) {
        if (msk[k2] != 0.f) {
          const v8h kr = Kv[k2 * 4];
          float acc = 0.f;
#pragma unroll
          for (int d = 0; d < 8; ++d) acc = fmaf(qv[d], (float)kr[d], acc);
          sr[k2] = acc * 0.35355339059327373f;   // 1/sqrt(8)
          mx = fmaxf(mx, sr[k2]);
        } else sr[k2] = -1e30f;
      }
      float ssum = 0.f;
#pragma unroll
      for (int k2 = 0; k2 < 9; ++k2) {
        float e = (msk[k2] != 0.f) ? __expf(sr[k2] - mx) : 0.f;
        sr[k2] = e; ssum += e;
      }
      const float inv = 1.f / ssum;   // vm token always unmasked -> ssum > 0
      float o[8] = {0.f, 0.f, 0.f, 0.f, 0.f, 0.f, 0.f, 0.f};
#pragma unroll
      for (int k2 = 0; k2 < 9; ++k2) {
        const float wgt = sr[k2] * inv;
        const v8h vr = Vv[k2 * 4];
#pragma unroll
        for (int d = 0; d < 8; ++d) o[d] = fmaf(wgt, (float)vr[d], o[d]);
      }
      // overwrite this (sample, head) slice of Q with O (exclusively owned)
      v8h orow;
#pragma unroll
      for (int d = 0; d < 8; ++d) orow[d] = (half_t)o[d];
      *(v8h*)(bq + (bb * 9 + q) * 32 + h * 8) = orow;
    }
  }

  // ---- x = O @ fc^T + tok (residual into tokf) ----
  {
    v16h bn0 = fragB(fcw, 0, lane), bn1 = fragB(fcw, 16, lane);
    for (int t = 0; t < 9; ++t) {
      v16h a = frag_f16(bq + t * 512, lane);
      v8f c0 = __builtin_amdgcn_wmma_f32_16x16x32_f16(false, a, false, bn0, (short)0, zero, false, false);
      v8f c1 = __builtin_amdgcn_wmma_f32_16x16x32_f16(false, a, false, bn1, (short)0, zero, false, false);
#pragma unroll
      for (int r = 0; r < 8; ++r) {
        const int row = t * 16 + mb + r;
        tokf[row * 32 + n]      += c0[r];
        tokf[row * 32 + 16 + n] += c1[r];
      }
    }
  }
  // ---- LayerNorm 1 (eps 1e-6, affine), in place ----
  for (int row = lane; row < 144; row += 32) {
    float* x = tokf + row * 32;
    float s = 0.f;
    for (int i = 0; i < 32; ++i) s += x[i];
    const float m = s * (1.f / 32.f);
    float v = 0.f;
    for (int i = 0; i < 32; ++i) { float d = x[i] - m; v = fmaf(d, d, v); }
    const float invs = rsqrtf(v * (1.f / 32.f) + 1e-6f);
    for (int i = 0; i < 32; ++i) x[i] = (x[i] - m) * invs * ln1g[i] + ln1b[i];
  }

  // ---- FFN: hidden = relu(x @ w1^T + b1) -> bk ----
  {
    v16h bn0 = fragB(w1, 0, lane), bn1 = fragB(w1, 16, lane);
    const float bias0 = bb1[n], bias1 = bb1[16 + n];
    for (int t = 0; t < 9; ++t) {
      v16h a = frag_f32(tokf + t * 512, lane);
      v8f c0 = __builtin_amdgcn_wmma_f32_16x16x32_f16(false, a, false, bn0, (short)0, zero, false, false);
      v8f c1 = __builtin_amdgcn_wmma_f32_16x16x32_f16(false, a, false, bn1, (short)0, zero, false, false);
#pragma unroll
      for (int r = 0; r < 8; ++r) {
        const int row = t * 16 + mb + r;
        bk[row * 32 + n]      = (half_t)fmaxf(c0[r] + bias0, 0.f);
        bk[row * 32 + 16 + n] = (half_t)fmaxf(c1[r] + bias1, 0.f);
      }
    }
  }
  // ---- y = hidden @ w2^T + b2 + x (residual into tokf) ----
  {
    v16h bn0 = fragB(w2, 0, lane), bn1 = fragB(w2, 16, lane);
    const float bias0 = bb2[n], bias1 = bb2[16 + n];
    for (int t = 0; t < 9; ++t) {
      v16h a = frag_f16(bk + t * 512, lane);
      v8f c0 = __builtin_amdgcn_wmma_f32_16x16x32_f16(false, a, false, bn0, (short)0, zero, false, false);
      v8f c1 = __builtin_amdgcn_wmma_f32_16x16x32_f16(false, a, false, bn1, (short)0, zero, false, false);
#pragma unroll
      for (int r = 0; r < 8; ++r) {
        const int row = t * 16 + mb + r;
        tokf[row * 32 + n]      += c0[r] + bias0;
        tokf[row * 32 + 16 + n] += c1[r] + bias1;
      }
    }
  }
  // ---- LayerNorm 2, in place ----
  for (int row = lane; row < 144; row += 32) {
    float* x = tokf + row * 32;
    float s = 0.f;
    for (int i = 0; i < 32; ++i) s += x[i];
    const float m = s * (1.f / 32.f);
    float v = 0.f;
    for (int i = 0; i < 32; ++i) { float d = x[i] - m; v = fmaf(d, d, v); }
    const float invs = rsqrtf(v * (1.f / 32.f) + 1e-6f);
    for (int i = 0; i < 32; ++i) x[i] = (x[i] - m) * invs * ln2g[i] + ln2b[i];
  }

  // ---- async store 144x32 f32 tokens LDS -> global ----
  {
    // async LDS reads are not ordered against prior DS writes: fence DScnt first
    asm volatile("s_wait_dscnt 0" ::: "memory");
    const unsigned ldsb = (unsigned)(size_t)tokf;
    const unsigned long long gbase = (unsigned long long)(size_t)gtok;
    for (int i = lane; i < (144 * 32) / 4; i += 32) {
      unsigned la = ldsb + (unsigned)i * 16u;
      unsigned long long ga = gbase + (unsigned long long)i * 16ull;
      asm volatile("global_store_async_from_lds_b128 %0, %1, off"
                   :: "v"(ga), "v"(la) : "memory");
    }
    asm volatile("s_wait_asynccnt 0" ::: "memory");
  }
}

// ---------------------------------------------------------------------------
// Autoencoder path
// ---------------------------------------------------------------------------
__global__ __launch_bounds__(256) void gather_sig(const float* __restrict__ tok,
                                                  float* __restrict__ s0)
{
  const int i = blockIdx.x * blockDim.x + threadIdx.x;
  if (i >= N_RAYSC * 32 * 128) return;
  const int l = i & 127, c = (i >> 7) & 31, nr = i >> 12;
  s0[i] = tok[(((size_t)(nr * 128 + l)) * 9 + 8) * 32 + c];
}

// conv1d(k=3, pad=1) + LN over L (eps 1e-5, no affine) + elu + optional maxpool2.
// One block per (n, cout) row, blockDim.x == L.  inB enables channel-concat.
__global__ void conv3_ln_elu(const float* __restrict__ inA, int Ca,
                             const float* __restrict__ inB, int Cb,
                             const float* __restrict__ w, const float* __restrict__ bias,
                             float* __restrict__ out, int Cout, int L, int do_pool)
{
  __shared__ float red[128];
  __shared__ float buf[128];
  const int nr = blockIdx.x / Cout, co = blockIdx.x % Cout;
  const int l = threadIdx.x;
  const int Cin = Ca + Cb;
  const float* wr = w + (size_t)co * Cin * 3;
  float acc = bias[co];
  for (int ci = 0; ci < Cin; ++ci) {
    const float* row = (ci < Ca) ? (inA + ((size_t)nr * Ca + ci) * L)
                                 : (inB + ((size_t)nr * Cb + (ci - Ca)) * L);
    const float* wk3 = wr + ci * 3;
    const float x0 = (l > 0)     ? row[l - 1] : 0.f;
    const float x1 = row[l];
    const float x2 = (l < L - 1) ? row[l + 1] : 0.f;
    acc = fmaf(wk3[0], x0, acc); acc = fmaf(wk3[1], x1, acc); acc = fmaf(wk3[2], x2, acc);
  }
  red[l] = acc; __syncthreads();
  for (int s = L >> 1; s > 0; s >>= 1) { if (l < s) red[l] += red[l + s]; __syncthreads(); }
  const float mean = red[0] / (float)L; __syncthreads();
  const float d0 = acc - mean;
  red[l] = d0 * d0; __syncthreads();
  for (int s = L >> 1; s > 0; s >>= 1) { if (l < s) red[l] += red[l + s]; __syncthreads(); }
  const float var = red[0] / (float)L;
  const float y = elu_f(d0 * rsqrtf(var + 1e-5f));
  if (do_pool) {
    buf[l] = y; __syncthreads();
    if (l < (L >> 1))
      out[((size_t)nr * Cout + co) * (L >> 1) + l] = fmaxf(buf[2 * l], buf[2 * l + 1]);
  } else {
    out[((size_t)nr * Cout + co) * L + l] = y;
  }
}

// transposed conv1d (stride 2, k=4, pad=1) + LN over Lout + elu.
// w layout (Cin, Cout, 4).  out[ol] = sum_ci sum_u w[ci][co][u] * x[(ol+1-u)/2].
__global__ void tconv_ln_elu(const float* __restrict__ inA, int Ca,
                             const float* __restrict__ inB, int Cb,
                             const float* __restrict__ w, const float* __restrict__ bias,
                             float* __restrict__ out, int Cout, int Lin)
{
  __shared__ float red[128];
  const int Lout = Lin * 2;
  const int nr = blockIdx.x / Cout, co = blockIdx.x % Cout;
  const int l = threadIdx.x;
  const int Cin = Ca + Cb;
  float acc = bias[co];
  for (int ci = 0; ci < Cin; ++ci) {
    const float* row = (ci < Ca) ? (inA + ((size_t)nr * Ca + ci) * Lin)
                                 : (inB + ((size_t)nr * Cb + (ci - Ca)) * Lin);
    const float* wk4 = w + ((size_t)ci * Cout + co) * 4;
    if ((l & 1) == 0) {
      acc = fmaf(wk4[1], row[l >> 1], acc);
      if (l >= 2) acc = fmaf(wk4[3], row[(l >> 1) - 1], acc);
    } else {
      const int i0 = (l + 1) >> 1;
      if (i0 < Lin) acc = fmaf(wk4[0], row[i0], acc);
      acc = fmaf(wk4[2], row[l >> 1], acc);
    }
  }
  red[l] = acc; __syncthreads();
  for (int s = Lout >> 1; s > 0; s >>= 1) { if (l < s) red[l] += red[l + s]; __syncthreads(); }
  const float mean = red[0] / (float)Lout; __syncthreads();
  const float d0 = acc - mean;
  red[l] = d0 * d0; __syncthreads();
  for (int s = Lout >> 1; s > 0; s >>= 1) { if (l < s) red[l] += red[l + s]; __syncthreads(); }
  const float var = red[0] / (float)Lout;
  out[((size_t)nr * Cout + co) * Lout + l] = elu_f(d0 * rsqrtf(var + 1e-5f));
}

// ---------------------------------------------------------------------------
// Heads
// ---------------------------------------------------------------------------
__global__ __launch_bounds__(256) void sigma_head(
    const float* __restrict__ cob,
    const float* __restrict__ s1w, const float* __restrict__ s1b,
    const float* __restrict__ s2w, const float* __restrict__ s2b,
    const float* __restrict__ s3w, const float* __restrict__ s3b,
    float* __restrict__ out)
{
  const int b = blockIdx.x * blockDim.x + threadIdx.x;
  if (b >= BTOT) return;
  const int nr = b >> 7, l = b & 127;
  float x[32];
#pragma unroll
  for (int c = 0; c < 32; ++c) x[c] = cob[((size_t)nr * 32 + c) * 128 + l];
  float h1[32];
  for (int o = 0; o < 32; ++o) {
    float acc = s1b[o];
    const float* wr = s1w + o * 32;
    for (int i = 0; i < 32; ++i) acc = fmaf(wr[i], x[i], acc);
    h1[o] = elu_f(acc);
  }
  float h2[16];
  for (int o = 0; o < 16; ++o) {
    float acc = s2b[o];
    const float* wr = s2w + o * 32;
    for (int i = 0; i < 32; ++i) acc = fmaf(wr[i], h1[i], acc);
    h2[o] = elu_f(acc);
  }
  float sg = s3b[0];
  for (int i = 0; i < 16; ++i) sg = fmaf(s3w[i], h2[i], sg);
  out[(size_t)b * 4 + 3] = fmaxf(sg, 0.f);
}

__global__ __launch_bounds__(128) void rgb_head(
    const float* __restrict__ tok, const float* __restrict__ vd,
    const float* __restrict__ feat, const float* __restrict__ masks,
    const float* __restrict__ r1w, const float* __restrict__ r1b,
    const float* __restrict__ r2w, const float* __restrict__ r2b,
    const float* __restrict__ r3w, const float* __restrict__ r3b,
    float* __restrict__ out)
{
  const int b = blockIdx.x * blockDim.x + threadIdx.x;
  if (b >= BTOT) return;
  float wv[8], mlocal[8];
  float mx = -1e30f; int any = 0;
  for (int v = 0; v < 8; ++v) {
    float x[41];
    const float* t = tok + ((size_t)b * 9 + v) * 32;
#pragma unroll
    for (int i = 0; i < 32; ++i) x[i] = t[i];
    const float* dv = vd + ((size_t)b * 8 + v) * 9;
#pragma unroll
    for (int i = 0; i < 9; ++i) x[32 + i] = dv[i];
    float h1[32];
    for (int o = 0; o < 32; ++o) {
      float acc = r1b[o];
      const float* wr = r1w + o * 41;
      for (int i = 0; i < 41; ++i) acc = fmaf(wr[i], x[i], acc);
      h1[o] = elu_f(acc);
    }
    float h2[16];
    for (int o = 0; o < 16; ++o) {
      float acc = r2b[o];
      const float* wr = r2w + o * 32;
      for (int i = 0; i < 32; ++i) acc = fmaf(wr[i], h1[i], acc);
      h2[o] = elu_f(acc);
    }
    float s = r3b[0];
    for (int i = 0; i < 16; ++i) s = fmaf(r3w[i], h2[i], s);
    const float m = masks[(size_t)b * 9 + v];
    mlocal[v] = m;
    if (m != 0.f) { wv[v] = s; if (s > mx) mx = s; any = 1; }
    else wv[v] = -1e30f;
  }
  float ssum = 0.f;
#pragma unroll
  for (int v = 0; v < 8; ++v) {
    float e = (mlocal[v] != 0.f) ? __expf(wv[v] - mx) : 0.f;
    wv[v] = e; ssum += e;
  }
  const float inv = any ? (1.f / ssum) : 0.f;
  float rgb0 = 0.f, rgb1 = 0.f, rgb2 = 0.f;
#pragma unroll
  for (int v = 0; v < 8; ++v) {
    const float ww = wv[v] * inv;
    const float* col = feat + ((size_t)b * 8 + v) * 36 + 32;
    rgb0 = fmaf(ww, col[0], rgb0);
    rgb1 = fmaf(ww, col[1], rgb1);
    rgb2 = fmaf(ww, col[2], rgb2);
  }
  out[(size_t)b * 4 + 0] = rgb0;
  out[(size_t)b * 4 + 1] = rgb1;
  out[(size_t)b * 4 + 2] = rgb2;
}

// ---------------------------------------------------------------------------
// Host launcher
// ---------------------------------------------------------------------------
extern "C" void kernel_launch(void* const* d_in, const int* in_sizes, int n_in,
                              void* d_out, int out_size, void* d_ws, size_t ws_size,
                              hipStream_t stream)
{
  (void)in_sizes; (void)n_in; (void)out_size; (void)ws_size;
  const float* viewdirs = (const float*)d_in[0];
  const float* feat     = (const float*)d_in[1];
  const float* occm     = (const float*)d_in[2];
  const float* tokw = (const float*)d_in[3];
  const float* tokb = (const float*)d_in[4];
  const float* vm1w = (const float*)d_in[5];
  const float* vm1b = (const float*)d_in[6];
  const float* vm2w = (const float*)d_in[7];
  const float* vm2b = (const float*)d_in[8];
  struct Lp { const float *wq,*wk,*wv,*fc,*g1,*be1,*w1,*b1,*w2,*b2,*g2,*be2; } L[4];
  for (int l = 0; l < 4; ++l) {
    const int p = 9 + l * 12;
    L[l].wq  = (const float*)d_in[p + 0];  L[l].wk  = (const float*)d_in[p + 1];
    L[l].wv  = (const float*)d_in[p + 2];  L[l].fc  = (const float*)d_in[p + 3];
    L[l].g1  = (const float*)d_in[p + 4];  L[l].be1 = (const float*)d_in[p + 5];
    L[l].w1  = (const float*)d_in[p + 6];  L[l].b1  = (const float*)d_in[p + 7];
    L[l].w2  = (const float*)d_in[p + 8];  L[l].b2  = (const float*)d_in[p + 9];
    L[l].g2  = (const float*)d_in[p + 10]; L[l].be2 = (const float*)d_in[p + 11];
  }
  int p = 57;
  const float* c1w = (const float*)d_in[p++]; const float* c1b = (const float*)d_in[p++];
  const float* c2w = (const float*)d_in[p++]; const float* c2b = (const float*)d_in[p++];
  const float* c3w = (const float*)d_in[p++]; const float* c3b = (const float*)d_in[p++];
  const float* t1w = (const float*)d_in[p++]; const float* t1bb = (const float*)d_in[p++];
  const float* t2w = (const float*)d_in[p++]; const float* t2bb = (const float*)d_in[p++];
  const float* t3w = (const float*)d_in[p++]; const float* t3bb = (const float*)d_in[p++];
  const float* cow = (const float*)d_in[p++]; const float* cobias = (const float*)d_in[p++];
  const float* s1w = (const float*)d_in[p++]; const float* s1b = (const float*)d_in[p++];
  const float* s2w = (const float*)d_in[p++]; const float* s2b = (const float*)d_in[p++];
  const float* s3w = (const float*)d_in[p++]; const float* s3b = (const float*)d_in[p++];
  const float* r1w = (const float*)d_in[p++]; const float* r1b = (const float*)d_in[p++];
  const float* r2w = (const float*)d_in[p++]; const float* r2b = (const float*)d_in[p++];
  const float* r3w = (const float*)d_in[p++]; const float* r3b = (const float*)d_in[p++];

  float* ws = (float*)d_ws;
  size_t off = 0;
  auto alloc = [&](size_t nElem) { float* q = ws + off; off += nElem; return q; };
  float* tok   = alloc((size_t)BTOT * 9 * 32);       // 151 MB
  float* masks = alloc((size_t)BTOT * 9);
  float* s0    = alloc((size_t)N_RAYSC * 32 * 128);  // autoencoder input (n,c,l)
  float* c1p   = alloc((size_t)N_RAYSC * 64 * 64);
  float* c2p   = alloc((size_t)N_RAYSC * 128 * 32);
  float* x3p   = alloc((size_t)N_RAYSC * 128 * 16);
  float* t1o   = alloc((size_t)N_RAYSC * 128 * 32);
  float* t2o   = alloc((size_t)N_RAYSC * 64 * 64);
  float* t3o   = alloc((size_t)N_RAYSC * 32 * 128);
  float* coo   = alloc((size_t)N_RAYSC * 32 * 128);
  float* out   = (float*)d_out;

  embed_kernel<<<BTOT / 256, 256, 0, stream>>>(feat, occm, tokw, tokb, vm1w, vm1b,
                                               vm2w, vm2b, tok, masks);
  for (int l = 0; l < 4; ++l)
    xf_layer<<<BTOT / 16, 32, 0, stream>>>(tok, masks,
        L[l].wq, L[l].wk, L[l].wv, L[l].fc, L[l].g1, L[l].be1,
        L[l].w1, L[l].b1, L[l].w2, L[l].b2, L[l].g2, L[l].be2);

  gather_sig<<<(N_RAYSC * 32 * 128) / 256, 256, 0, stream>>>(tok, s0);
  conv3_ln_elu<<<N_RAYSC * 64,  128, 0, stream>>>(s0,  32, s0,  0,   c1w, c1b, c1p, 64, 128, 1);
  conv3_ln_elu<<<N_RAYSC * 128,  64, 0, stream>>>(c1p, 64, c1p, 0,   c2w, c2b, c2p, 128, 64, 1);
  conv3_ln_elu<<<N_RAYSC * 128,  32, 0, stream>>>(c2p, 128, c2p, 0,  c3w, c3b, x3p, 128, 32, 1);
  tconv_ln_elu<<<N_RAYSC * 128,  32, 0, stream>>>(x3p, 128, x3p, 0,  t1w, t1bb, t1o, 128, 16);
  tconv_ln_elu<<<N_RAYSC * 64,   64, 0, stream>>>(t1o, 128, c2p, 128, t2w, t2bb, t2o, 64, 32);
  tconv_ln_elu<<<N_RAYSC * 32,  128, 0, stream>>>(t2o, 64,  c1p, 64,  t3w, t3bb, t3o, 32, 64);
  conv3_ln_elu<<<N_RAYSC * 32,  128, 0, stream>>>(t3o, 32,  s0,  32,  cow, cobias, coo, 32, 128, 0);

  sigma_head<<<BTOT / 256, 256, 0, stream>>>(coo, s1w, s1b, s2w, s2b, s3w, s3b, out);
  rgb_head<<<BTOT / 128, 128, 0, stream>>>(tok, viewdirs, feat, masks,
                                           r1w, r1b, r2w, r2b, r3w, r3b, out);
}
